// JointMIEstimator_14422500180356
// MI455X (gfx1250) — matrix operand order repaired and checked by reading, not compiled, not run on hardware
//
#include <hip/hip_runtime.h>
#include <hip/hip_bf16.h>
#include <math.h>

#define B_SZ   1024
#define D_SZ   768
#define HID    384
#define MLP_H  300
#define NPAD   304      // 19 tiles of 16
#define LD_AB  304      // leading dim of Ax / By
#define KC     60       // k-chunk for T1 tiling (5 * 60 = 300)
#define LDSP   68       // padded row stride for T1 LDS tiles (272B, 16B aligned)

typedef __attribute__((ext_vector_type(16))) _Float16 v16h;
typedef __attribute__((ext_vector_type(8)))  _Float16 v8h;
typedef __attribute__((ext_vector_type(8)))  float    v8f;
typedef __attribute__((ext_vector_type(4)))  float    v4f;

__device__ __forceinline__ float softplusf(float x) {
    if (x > 20.0f)  return x;
    if (x < -20.0f) return expf(x);
    return log1pf(expf(x));
}

// ---------------------------------------------------------------------------
// Kernel 0: convert W1 (300x768 f32) -> W1h (304x768 f16), zero-pad rows 300..303
// ---------------------------------------------------------------------------
__global__ void k_convW(const float* __restrict__ W1, _Float16* __restrict__ W1h) {
    int idx = blockIdx.x * 256 + threadIdx.x;          // [0, 304*768)
    int n = idx / D_SZ;
    W1h[idx] = (n < MLP_H) ? (_Float16)W1[idx] : (_Float16)0.0f;
}

// ---------------------------------------------------------------------------
// Kernel 1: z1/z2 weighted reductions, softplus, reparam samples (f16 out),
//           symmetric-KL partial sum (log-sigma terms cancel analytically).
// One thread per (b, h). grid = B*HID/256.
// ---------------------------------------------------------------------------
__global__ void k_stage1(const float* __restrict__ zl, const float* __restrict__ zv,
                         const float* __restrict__ wl, const float* __restrict__ bl,
                         const float* __restrict__ wv, const float* __restrict__ bv,
                         const float* __restrict__ eps1, const float* __restrict__ eps2,
                         _Float16* __restrict__ s1h, _Float16* __restrict__ s2h,
                         float* __restrict__ kl_acc) {
    int idx = blockIdx.x * 256 + threadIdx.x;          // b*HID + h
    int b = idx / HID;
    int h = idx - b * HID;

    const float* zlb = zl + (size_t)b * 20 * D_SZ;
    const float* zvb = zv + (size_t)b * 36 * D_SZ;

    float z1lo = 0.f, z1hi = 0.f;
#pragma unroll 4
    for (int s = 0; s < 20; ++s) {
        float w = wl[s];
        z1lo += zlb[s * D_SZ + h]       * w;
        z1hi += zlb[s * D_SZ + h + HID] * w;
    }
    z1lo += bl[0]; z1hi += bl[0];

    float z2lo = 0.f, z2hi = 0.f;
#pragma unroll 4
    for (int s = 0; s < 36; ++s) {
        float w = wv[s];
        z2lo += zvb[s * D_SZ + h]       * w;
        z2hi += zvb[s * D_SZ + h + HID] * w;
    }
    z2lo += bv[0]; z2hi += bv[0];

    float mu1 = z1lo, sig1 = softplusf(z1hi) + 1e-7f;
    float mu2 = z2lo, sig2 = softplusf(z2hi) + 1e-7f;
    float e1 = eps1[idx], e2 = eps2[idx];
    float s1 = mu1 + sig1 * e1;
    float s2 = mu2 + sig2 * e2;

    s1h[idx] = (_Float16)s1;
    s2h[idx] = (_Float16)s2;

    // (kl_1_2 + kl_2_1) per element: log-sigma & const terms cancel.
    float q12 = (s1 - mu2) / sig2;
    float q21 = (s2 - mu1) / sig1;
    float t = 0.5f * (q12 * q12 - e1 * e1 + q21 * q21 - e2 * e2);

    __shared__ float red[256];
    red[threadIdx.x] = t;
    __syncthreads();
    for (int s = 128; s > 0; s >>= 1) {
        if (threadIdx.x < s) red[threadIdx.x] += red[threadIdx.x + s];
        __syncthreads();
    }
    if (threadIdx.x == 0) atomicAdd(kl_acc, red[0]);
}

// ---------------------------------------------------------------------------
// Kernel 2: WMMA GEMMs.  Ax = s1 @ W1x^T (+ b1 folded), By = s2 @ W1y^T.
// M=1024, N=304 (19 tiles), K=384 (12 steps of 32). One 16x16 tile per wave.
// A fragment (16x32 f16): lane = 16*half + m; elems 0..7  -> K = 16*half+0..7
//                                             elems 8..15 -> K = 16*half+8+0..7
// B fragment (32x16 f16): lane = 16*half + n; elem e -> K = 16*half + e
// C/D (16x16 f32, 8 VGPRs): value r -> row = r + 8*half, col = lane&15
// ---------------------------------------------------------------------------
__global__ void k_gemm(const _Float16* __restrict__ s1h, const _Float16* __restrict__ s2h,
                       const _Float16* __restrict__ W1h, const float* __restrict__ b1,
                       float* __restrict__ Ax, float* __restrict__ By) {
    int wave = threadIdx.x >> 5;
    int lane = threadIdx.x & 31;
    int tile = blockIdx.x * 8 + wave;                  // [0, 2432)
    int gemm = tile / (64 * 19);                       // 0 -> Ax, 1 -> By
    int rem  = tile - gemm * (64 * 19);
    int mt = rem / 19;
    int nt = rem - mt * 19;

    const _Float16* A  = gemm ? s2h : s1h;
    float* OUT         = gemm ? By  : Ax;

    int half = lane >> 4;
    int m    = lane & 15;
    const _Float16* Abase = A   + (size_t)(mt * 16 + m) * HID  + half * 16;
    const _Float16* Bbase = W1h + (size_t)(nt * 16 + m) * D_SZ + gemm * HID + half * 16;

    v8f c = {};
#pragma unroll
    for (int kc = 0; kc < 12; ++kc) {
        int k0 = kc * 32;
        v8h alo = *(const v8h*)(Abase + k0);           // K = k0 + 16*half + 0..7
        v8h ahi = *(const v8h*)(Abase + k0 + 8);       // K = k0 + 16*half + 8 + 0..7
        v16h a;
#pragma unroll
        for (int i = 0; i < 8; ++i) { a[i] = alo[i]; a[i + 8] = ahi[i]; }
        v16h bf = *(const v16h*)(Bbase + k0);          // K = k0 + 16*half + 0..15
        c = __builtin_amdgcn_wmma_f32_16x16x32_f16(
                /*neg_a=*/false, a, /*neg_b=*/false, bf,
                /*c_mod=*/(short)0, c, /*reuse_a=*/false, /*reuse_b=*/false);
    }

    int col  = nt * 16 + (lane & 15);
    float bias = (gemm == 0 && col < MLP_H) ? b1[col] : 0.0f;
#pragma unroll
    for (int r = 0; r < 8; ++r) {
        int row = mt * 16 + r + 8 * half;
        OUT[(size_t)row * LD_AB + col] = c[r] + bias;
    }
}

// ---------------------------------------------------------------------------
// Kernel 3: T0 mean.  t0(b) = softplus( sum_k relu(Ax+By)*w2 + b2 )
// (b1 already folded into Ax). One wave per row b -> coalesced lane-strided reads.
// ---------------------------------------------------------------------------
__global__ void k_T0(const float* __restrict__ Ax, const float* __restrict__ By,
                     const float* __restrict__ W2, const float* __restrict__ b2,
                     float* __restrict__ t0_acc) {
    int wave = threadIdx.x >> 5;
    int lane = threadIdx.x & 31;
    int b = blockIdx.x * 8 + wave;
    const float* ax = Ax + (size_t)b * LD_AB;
    const float* by = By + (size_t)b * LD_AB;
    float acc = 0.f;
    for (int k = lane; k < MLP_H; k += 32)
        acc += fmaxf(ax[k] + by[k], 0.f) * W2[k];
    acc += __shfl_xor(acc, 16, 32);
    acc += __shfl_xor(acc, 8, 32);
    acc += __shfl_xor(acc, 4, 32);
    acc += __shfl_xor(acc, 2, 32);
    acc += __shfl_xor(acc, 1, 32);
    if (lane == 0) atomicAdd(t0_acc, softplusf(acc + b2[0]));
}

// ---------------------------------------------------------------------------
// Kernel 4: T1 exp-sums. Block = 64 q-cols x 64 p-rows, 4x4 register micro-tile.
// LDS tiles stored k-major ([k][row], row stride 68 floats) so each thread's
// 4 q-values / 4 p-values are one ds_load_b128 each (2 LDS ops : 48 VALU ops).
// Next k-chunk is prefetched (global_prefetch_b8) while current one computes.
// ---------------------------------------------------------------------------
__global__ void k_T1(const float* __restrict__ Ax, const float* __restrict__ By,
                     const float* __restrict__ W2, const float* __restrict__ b2,
                     const int* __restrict__ perm, float* __restrict__ S) {
    __shared__ __align__(16) float Axs[KC][LDSP];   // [k][q]
    __shared__ __align__(16) float Bys[KC][LDSP];   // [k][p]
    __shared__ float w2s[KC];
    __shared__ int   prm[64];

    int tx = threadIdx.x & 15;     // q micro-group
    int ty = threadIdx.x >> 4;     // p micro-group
    int q0 = blockIdx.x * 64;
    int p0 = blockIdx.y * 64;

    if (threadIdx.x < 64) prm[threadIdx.x] = perm[p0 + threadIdx.x];
    __syncthreads();

    float acc[4][4] = {};          // [p_local][q_local]

    for (int kc = 0; kc < 5; ++kc) {
        int kb = kc * KC;
        // stage tiles transposed: consecutive threads walk k (coalesced global
        // reads), LDS writes scatter with stride LDSP (mild bank rotation).
        for (int t = threadIdx.x; t < 64 * KC; t += 256) {
            int r = t / KC, c = t - r * KC;
            Axs[c][r] = Ax[(size_t)(q0 + r) * LD_AB + kb + c];
            Bys[c][r] = By[(size_t)prm[r]  * LD_AB + kb + c];
        }
        if (threadIdx.x < KC) w2s[threadIdx.x] = W2[kb + threadIdx.x];
        // prefetch next chunk into cache while this one computes
        if (kc < 4) {
            int r = threadIdx.x & 63;
            __builtin_prefetch(&Ax[(size_t)(q0 + r) * LD_AB + kb + KC], 0, 1);
            __builtin_prefetch(&By[(size_t)prm[r]  * LD_AB + kb + KC], 0, 1);
        }
        __syncthreads();

        for (int k = 0; k < KC; ++k) {
            float w = w2s[k];
            v4f aq = *(const v4f*)&Axs[k][tx * 4];
            v4f bp = *(const v4f*)&Bys[k][ty * 4];
#pragma unroll
            for (int i = 0; i < 4; ++i)
#pragma unroll
                for (int j = 0; j < 4; ++j)
                    acc[i][j] += fmaxf(bp[i] + aq[j], 0.f) * w;
        }
        __syncthreads();
    }

    float b2v = b2[0];
#pragma unroll
    for (int i = 0; i < 4; ++i) {
        float es = 0.f;
#pragma unroll
        for (int j = 0; j < 4; ++j)
            es += expf(softplusf(acc[i][j] + b2v));
        // reduce over the 16 tx lanes (xor of bits 0..3 stays inside the group)
        es += __shfl_xor(es, 1, 32);
        es += __shfl_xor(es, 2, 32);
        es += __shfl_xor(es, 4, 32);
        es += __shfl_xor(es, 8, 32);
        if (tx == 0) atomicAdd(&S[p0 + ty * 4 + i], es);
    }
}

// ---------------------------------------------------------------------------
// Kernel 5: finalize.  out = d_skl - [ T0_mean - (mean_p log S[p] - log B) ]
// ---------------------------------------------------------------------------
__global__ void k_final(const float* __restrict__ S, const float* __restrict__ kl_acc,
                        const float* __restrict__ t0_acc, float* __restrict__ out) {
    __shared__ float red[256];
    float s = 0.f;
    for (int p = threadIdx.x; p < B_SZ; p += 256) s += logf(S[p]);
    red[threadIdx.x] = s;
    __syncthreads();
    for (int k = 128; k > 0; k >>= 1) {
        if (threadIdx.x < k) red[threadIdx.x] += red[threadIdx.x + k];
        __syncthreads();
    }
    if (threadIdx.x == 0) {
        float lse_mean = red[0] / (float)B_SZ;
        float d_skl    = kl_acc[0] / (2.0f * (float)B_SZ);
        float t0_mean  = t0_acc[0] / (float)B_SZ;
        float lower    = t0_mean - (lse_mean - logf((float)B_SZ));
        out[0] = d_skl - lower;
    }
}

// ---------------------------------------------------------------------------
extern "C" void kernel_launch(void* const* d_in, const int* in_sizes, int n_in,
                              void* d_out, int out_size, void* d_ws, size_t ws_size,
                              hipStream_t stream) {
    const float* zl   = (const float*)d_in[0];
    const float* zv   = (const float*)d_in[1];
    const float* wl   = (const float*)d_in[2];
    const float* bl   = (const float*)d_in[3];
    const float* wv   = (const float*)d_in[4];
    const float* bv   = (const float*)d_in[5];
    const float* W1   = (const float*)d_in[6];
    const float* b1   = (const float*)d_in[7];
    const float* W2   = (const float*)d_in[8];
    const float* b2   = (const float*)d_in[9];
    const float* eps1 = (const float*)d_in[10];
    const float* eps2 = (const float*)d_in[11];
    const int*   perm = (const int*)d_in[12];
    float* out = (float*)d_out;

    char* ws = (char*)d_ws;
    size_t off = 0;
    auto alloc = [&](size_t bytes) -> void* {
        void* p = ws + off;
        off += (bytes + 255) & ~(size_t)255;
        return p;
    };

    // accumulators first (single contiguous memset region)
    float* kl_acc = (float*)alloc(4);                         // 256 B slot
    float* t0_acc = (float*)alloc(4);                         // 256 B slot
    float* S      = (float*)alloc(B_SZ * sizeof(float));      // 4096 B
    size_t zero_bytes = off;

    _Float16* s1h = (_Float16*)alloc((size_t)B_SZ * HID * sizeof(_Float16));
    _Float16* s2h = (_Float16*)alloc((size_t)B_SZ * HID * sizeof(_Float16));
    _Float16* W1h = (_Float16*)alloc((size_t)NPAD * D_SZ * sizeof(_Float16));
    float*    Ax  = (float*)alloc((size_t)B_SZ * LD_AB * sizeof(float));
    float*    By  = (float*)alloc((size_t)B_SZ * LD_AB * sizeof(float));

    hipMemsetAsync(d_ws, 0, zero_bytes, stream);

    k_convW <<< (NPAD * D_SZ) / 256, 256, 0, stream >>> (W1, W1h);
    k_stage1<<< (B_SZ * HID) / 256,  256, 0, stream >>> (zl, zv, wl, bl, wv, bv,
                                                         eps1, eps2, s1h, s2h, kl_acc);
    k_gemm  <<< 304,                 256, 0, stream >>> (s1h, s2h, W1h, b1, Ax, By);
    k_T0    <<< B_SZ / 8 / 32,       256, 0, stream >>> (Ax, By, W2, b2, t0_acc);
    k_T1    <<< dim3(16, 16),        256, 0, stream >>> (Ax, By, W2, b2, perm, S);
    k_final <<< 1,                   256, 0, stream >>> (S, kl_acc, t0_acc, out);
}
